// LSTMScaledDotProductAttentionNetwork_27702539059325
// MI455X (gfx1250) — matrix operand order repaired and checked by reading, not compiled, not run on hardware
//
#include <hip/hip_runtime.h>
#include <stdint.h>
#include <stddef.h>

// ---------------- problem constants (from reference) ----------------
#define BB    64      // batch
#define SS    512     // seq len
#define EE    300     // embedding dim
#define EP0   320     // embedding dim padded to /32
#define HH    256     // hidden
#define NG    1024    // 4*H gate width
#define D2    512     // 2*H
#define SENTD 64      // sentiment dim
#define QIND  576     // 2H + SENT

typedef __attribute__((ext_vector_type(16))) __bf16 v16bf;
typedef __attribute__((ext_vector_type(8)))  float  v8f;

union FragBF { v16bf v; uint4 q[2]; };

__device__ __forceinline__ unsigned short f2bf(float f) {
  unsigned u = __float_as_uint(f);
  unsigned r = u + 0x7FFFu + ((u >> 16) & 1u);   // round-to-nearest-even
  return (unsigned short)(r >> 16);
}
__device__ __forceinline__ float bf2f(unsigned short h) {
  return __uint_as_float(((unsigned)h) << 16);
}
__device__ __forceinline__ float sigm(float x) { return 1.0f / (1.0f + __expf(-x)); }

__device__ __forceinline__ v8f wmma_bf16(v16bf a, v16bf b, v8f c) {
  return __builtin_amdgcn_wmma_f32_16x16x32_bf16(false, a, false, b, (short)0, c, false, false);
}

// ---- CDNA5 async copy: global -> LDS, 16B per lane, tracked by ASYNCcnt ----
__device__ __forceinline__ void async_b128(unsigned lds_byte_off, const void* gaddr) {
  asm volatile("global_load_async_to_lds_b128 %0, %1, off"
               :: "v"(lds_byte_off), "v"((unsigned long long)(uintptr_t)gaddr)
               : "memory");
}
__device__ __forceinline__ void wait_async0() {
  asm volatile("s_wait_asynccnt 0x0" ::: "memory");
}
// LDS byte offset = low 32 bits of the generic pointer (flat aperture rule)
__device__ __forceinline__ unsigned lds_off(const void* p) {
  return (unsigned)(uintptr_t)p;
}

// A fragment from a row-major bf16 panel (ld elements per row), 16x32 tile at (m0,k0)
__device__ __forceinline__ v16bf frag_a(const unsigned short* base, int ld, int m0, int k0, int lane) {
  int m  = m0 + (lane & 15);
  int kb = k0 + ((lane >> 4) << 3);
  const unsigned short* p = base + (size_t)m * ld + kb;
  FragBF f;
  f.q[0] = *reinterpret_cast<const uint4*>(p);
  f.q[1] = *reinterpret_cast<const uint4*>(p + 16);
  return f.v;
}
// B fragment from Bt[N,K] (NT layout), 32x16 tile at (k0, n0)
__device__ __forceinline__ v16bf frag_b(const unsigned short* Bt, int ldb, int n0, int k0, int lane) {
  int n  = n0 + (lane & 15);
  int kb = k0 + ((lane >> 4) << 3);
  const unsigned short* p = Bt + (size_t)n * ldb + kb;
  FragBF f;
  f.q[0] = *reinterpret_cast<const uint4*>(p);
  f.q[1] = *reinterpret_cast<const uint4*>(p + 16);
  return f.v;
}

// ---------------- weight / data prep kernels ----------------

__global__ void prep_lstm_w_kernel(const float* __restrict__ Wih, const float* __restrict__ Whh,
                                   unsigned short* __restrict__ W2, int din, int epl) {
  int KD = epl + HH;
  int n = blockIdx.y;
  int k = blockIdx.x * 256 + threadIdx.x;
  if (k >= KD) return;
  float v;
  if (k < din)      v = Wih[(size_t)n * din + k];
  else if (k < epl) v = 0.0f;
  else              v = Whh[(size_t)n * HH + (k - epl)];
  W2[(size_t)n * KD + k] = f2bf(v);
}

__global__ void prep_bias_kernel(const float* __restrict__ bih, const float* __restrict__ bhh,
                                 float* __restrict__ bout) {
  int i = blockIdx.x * 256 + threadIdx.x;
  if (i < NG) bout[i] = bih[i] + bhh[i];
}

__global__ void f32_to_bf16_kernel(const float* __restrict__ src, unsigned short* __restrict__ dst,
                                   long long n) {
  long long i = (long long)blockIdx.x * 256 + threadIdx.x;
  if (i < n) dst[i] = f2bf(src[i]);
}

__global__ void embed_kernel(const int* __restrict__ tok, const float* __restrict__ emb,
                             unsigned short* __restrict__ x0) {
  long long i = (long long)blockIdx.x * 256 + threadIdx.x;
  if (i >= (long long)BB * SS * EP0) return;
  long long row = i / EP0;
  int k = (int)(i - row * EP0);
  int t = tok[row];
  float v = (k < EE) ? emb[(size_t)t * EE + k] : 0.0f;
  x0[i] = f2bf(v);
}

__global__ void transpose_kernel(const unsigned short* __restrict__ enc,
                                 unsigned short* __restrict__ encT) {
  long long i = (long long)blockIdx.x * 256 + threadIdx.x;
  if (i >= (long long)BB * D2 * SS) return;
  int b = (int)(i >> 18);
  int rem = (int)(i & 262143);
  int d = rem >> 9;
  int s = rem & 511;
  encT[i] = enc[((long long)b * SS + s) * D2 + d];
}

__global__ void qin_kernel(const unsigned short* __restrict__ enc, const float* __restrict__ sent,
                           unsigned short* __restrict__ qin) {
  long long i = (long long)blockIdx.x * 256 + threadIdx.x;
  if (i >= (long long)BB * SS * QIND) return;
  long long row = i / QIND;
  int c = (int)(i - row * QIND);
  int b = (int)(row >> 9);
  qin[i] = (c < D2) ? enc[row * D2 + c] : f2bf(sent[b * SENTD + (c - D2)]);
}

// ---------------- NT GEMM with double-buffered async-LDS staging ----------------
// C[M,N] = A[M,K]bf16 * Bt[N,K]bf16^T.  Block = 256 thr = 8 waves (4m x 2n),
// block tile 128x64, wave tile 32x32.  A/B k-panels staged in LDS via
// GLOBAL_LOAD_ASYNC_TO_LDS_B128, ping-pong, one barrier per k-step.
// Requires M%128==0, N%64==0, K%32==0 (true for all launches here).
// mode 0: Cf = acc*alpha + bias (f32);  mode 1: Cbf = bf16(tanh(acc*alpha+bias))
__global__ __launch_bounds__(256) void gemm_nt_kernel(
    const unsigned short* __restrict__ A, int lda, long long strideA,
    const unsigned short* __restrict__ Bt, int ldb, long long strideB,
    float* __restrict__ Cf, unsigned short* __restrict__ Cbf, int ldc, long long strideC,
    int M, int N, int K, int mode, float alpha, const float* __restrict__ bias) {
  __shared__ unsigned short ldsA[2][128 * 32];   // 2 x 8 KB
  __shared__ unsigned short ldsB[2][64 * 32];    // 2 x 4 KB
  const int lane = threadIdx.x & 31;
  const int wave = threadIdx.x >> 5;
  const int wm = wave >> 1, wn = wave & 1;
  const int m0blk = blockIdx.y * 128;
  const int n0blk = blockIdx.x * 64;
  const long long bz = blockIdx.z;
  A  += bz * strideA;
  Bt += bz * strideB;

  // async issue of one k-panel into stage s (no divergence: exact chunk counts)
  auto issue = [&](int s, int k0) {
    // A panel: 128 rows x 64B = 512 chunks, 2 per thread
    {
      int c = threadIdx.x;
      int mr = c >> 2, ch = c & 3;
      async_b128(lds_off(&ldsA[s][0]) + mr * 64 + ch * 16,
                 A + (size_t)(m0blk + mr) * lda + k0 + ch * 8);
      c += 256; mr = c >> 2; ch = c & 3;
      async_b128(lds_off(&ldsA[s][0]) + mr * 64 + ch * 16,
                 A + (size_t)(m0blk + mr) * lda + k0 + ch * 8);
    }
    // B panel: 64 rows x 64B = 256 chunks, 1 per thread
    {
      int c = threadIdx.x;
      int nr = c >> 2, ch = c & 3;
      async_b128(lds_off(&ldsB[s][0]) + nr * 64 + ch * 16,
                 Bt + (size_t)(n0blk + nr) * ldb + k0 + ch * 8);
    }
  };

  int s = 0;
  issue(0, 0);
  wait_async0();
  __syncthreads();

  v8f acc[2][2] = {};
  for (int k0 = 0; k0 < K; k0 += 32) {
    const bool more = (k0 + 32) < K;
    if (more) issue(s ^ 1, k0 + 32);
    v16bf a0 = frag_a(&ldsA[s][0], 32, wm * 32,      0, lane);
    v16bf a1 = frag_a(&ldsA[s][0], 32, wm * 32 + 16, 0, lane);
    v16bf b0 = frag_b(&ldsB[s][0], 32, wn * 32,      0, lane);
    v16bf b1 = frag_b(&ldsB[s][0], 32, wn * 32 + 16, 0, lane);
    acc[0][0] = wmma_bf16(a0, b0, acc[0][0]);
    acc[0][1] = wmma_bf16(a0, b1, acc[0][1]);
    acc[1][0] = wmma_bf16(a1, b0, acc[1][0]);
    acc[1][1] = wmma_bf16(a1, b1, acc[1][1]);
    if (more) {
      wait_async0();
      __syncthreads();
      s ^= 1;
    }
  }

#pragma unroll
  for (int im = 0; im < 2; im++) {
#pragma unroll
    for (int in = 0; in < 2; in++) {
      int n = n0blk + wn * 32 + in * 16 + (lane & 15);
      float bv = bias ? bias[n] : 0.0f;
#pragma unroll
      for (int r = 0; r < 8; r++) {
        int m = m0blk + wm * 32 + im * 16 + r + ((lane >> 4) << 3);
        float v = acc[im][in][r] * alpha + bv;
        long long idx = bz * strideC + (long long)m * ldc + n;
        if (mode == 1) Cbf[idx] = f2bf(tanhf(v));
        else           Cf[idx]  = v;
      }
    }
  }
}

// ---------------- persistent bidirectional LSTM layer ----------------
// blockIdx.x = direction.  1024 thr = 32 waves.  Per step:
//   gates[64,1024] = [x_t(pad) | h] @ W2^T  via WMMA.
// h ping-pongs in LDS, c lives in VGPRs, and x_{t+1} is prefetched into LDS
// with async copies *during* step t so only h sits on the recurrence chain.
// Dynamic LDS: hbuf 2*32KB + xbuf 2*(64*epl*2)B.
// NOTE: LDS pointers are computed arithmetically (never stored in initialized
// pointer arrays) to avoid addrspacecast static initializers.
__global__ __launch_bounds__(1024, 1) void lstm_kernel(
    const unsigned short* __restrict__ xin, int ldx, int epl,
    const unsigned short* __restrict__ W2all,   // [2][NG][epl+HH] bf16
    const float* __restrict__ biasAll,          // [2][NG]
    const int* __restrict__ lens,
    unsigned short* __restrict__ out) {         // [B*S][D2] bf16, dir picks col half
  extern __shared__ unsigned short smem[];
  const int dir = blockIdx.x;
  const int KD = epl + HH;
  const unsigned short* W2 = W2all + (size_t)dir * NG * KD;
  const float* bias = biasAll + (size_t)dir * NG;
  const int lane = threadIdx.x & 31;
  const int wave = threadIdx.x >> 5;
  const int cpr = epl >> 3;              // 16B chunks per x row

#define HBUF(s) (smem + (s) * (BB * HH))
#define XBUF(s) (smem + 2 * (BB * HH) + (s) * (BB * epl))

  // async prefetch of the x panel for time step t into x stage qs
  auto issue_x = [&](int qs, int t) {
    unsigned xbase = lds_off(XBUF(qs));
    for (int c = threadIdx.x; c < BB * cpr; c += 1024) {
      int row = c / cpr, ch = c - row * cpr;
      async_b128(xbase + row * epl * 2 + ch * 16,
                 xin + ((size_t)row * SS + t) * ldx + ch * 8);
    }
  };

  for (int i = threadIdx.x; i < BB * HH; i += 1024) { HBUF(0)[i] = 0; HBUF(1)[i] = 0; }
  issue_x(0, dir ? (SS - 1) : 0);
  wait_async0();
  __syncthreads();

  v8f cst[2] = {};   // persistent cell-state tiles (one per gate-group)
  int p = 0, q = 0;
  for (int step = 0; step < SS; step++) {
    int t = dir ? (SS - 1 - step) : step;
    if (step + 1 < SS) issue_x(q ^ 1, dir ? (SS - 2 - step) : (step + 1));
    const unsigned short* hcur = HBUF(p);
    unsigned short*       hnew = HBUF(p ^ 1);
    const unsigned short* xcur = XBUF(q);
#pragma unroll
    for (int gg = 0; gg < 2; gg++) {
      int gi = wave * 2 + gg;          // 0..63
      int m0 = (gi & 3) * 16;          // batch tile
      int j0 = (gi >> 2) * 16;         // hidden-column tile
      v8f acc[4] = {};                 // i, f, g, o
      for (int k0 = 0; k0 < KD; k0 += 32) {
        v16bf a;
        int m  = m0 + (lane & 15);
        int kb = ((lane >> 4) << 3);
        if (k0 < epl) {                // x_t from LDS prefetch buffer
          const unsigned short* pA = xcur + (size_t)m * epl + k0 + kb;
          FragBF f;
          f.q[0] = *reinterpret_cast<const uint4*>(pA);
          f.q[1] = *reinterpret_cast<const uint4*>(pA + 16);
          a = f.v;
        } else {                       // h from LDS ping-pong
          const unsigned short* pH = hcur + (size_t)m * HH + (k0 - epl) + kb;
          FragBF f;
          f.q[0] = *reinterpret_cast<const uint4*>(pH);
          f.q[1] = *reinterpret_cast<const uint4*>(pH + 16);
          a = f.v;
        }
#pragma unroll
        for (int g = 0; g < 4; g++) {
          v16bf b = frag_b(W2, KD, g * HH + j0, k0, lane);
          acc[g] = wmma_bf16(a, b, acc[g]);
        }
      }
      // cell update: all in registers; reads old h from hcur, writes hnew
      int j = j0 + (lane & 15);
      float b0v = bias[j], b1v = bias[HH + j], b2v = bias[2 * HH + j], b3v = bias[3 * HH + j];
#pragma unroll
      for (int r = 0; r < 8; r++) {
        int m = m0 + r + ((lane >> 4) << 3);
        bool mt = t < lens[m];
        float iv = sigm(acc[0][r] + b0v);
        float fv = sigm(acc[1][r] + b1v);
        float gv = tanhf(acc[2][r] + b2v);
        float ov = sigm(acc[3][r] + b3v);
        float c_old = cst[gg][r];
        float c_n = fv * c_old + iv * gv;
        float h_n = ov * tanhf(c_n);
        float h_old = bf2f(hcur[m * HH + j]);
        cst[gg][r] = mt ? c_n : c_old;
        hnew[m * HH + j] = f2bf(mt ? h_n : h_old);
        out[((size_t)m * SS + t) * D2 + dir * HH + j] = f2bf(mt ? h_n : 0.0f);
      }
    }
    wait_async0();
    __syncthreads();
    p ^= 1;
    q ^= 1;
  }
#undef HBUF
#undef XBUF
}

// ---------------- softmax over 512-wide rows (one wave32 per row) ----------------
__global__ __launch_bounds__(256) void softmax_kernel(const float* __restrict__ scores,
                                                      const int* __restrict__ lens,
                                                      unsigned short* __restrict__ attn) {
  int lane = threadIdx.x & 31, wave = threadIdx.x >> 5;
  int row = blockIdx.x * 8 + wave;
  int b = row >> 9;
  int len = lens[b];
  const float scale = 0.0441941738241592f;         // 1/sqrt(512)
  const float* src = scores + (long long)row * SS;
  float vals[16];
  float mx = -1e30f;
#pragma unroll
  for (int i = 0; i < 16; i++) {
    int k = lane + i * 32;
    float v = (k < len) ? src[k] * scale : -1e30f;
    vals[i] = v;
    mx = fmaxf(mx, v);
  }
  for (int off = 16; off; off >>= 1) mx = fmaxf(mx, __shfl_xor(mx, off, 32));
  float sum = 0.0f;
#pragma unroll
  for (int i = 0; i < 16; i++) { vals[i] = __expf(vals[i] - mx); sum += vals[i]; }
  for (int off = 16; off; off >>= 1) sum += __shfl_xor(sum, off, 32);
  float inv = 1.0f / sum;
  unsigned short* dst = attn + (long long)row * SS;
#pragma unroll
  for (int i = 0; i < 16; i++) dst[lane + i * 32] = f2bf(vals[i] * inv);
}

// ---------------- final readout: sigmoid(mask(attended) . Wp + bp) ----------------
__global__ __launch_bounds__(256) void final_kernel(const float* __restrict__ attended,
                                                    const float* __restrict__ Wp,
                                                    const float* __restrict__ bp,
                                                    const int* __restrict__ lens,
                                                    float* __restrict__ out) {
  int lane = threadIdx.x & 31, wave = threadIdx.x >> 5;
  int row = blockIdx.x * 8 + wave;
  int b = row >> 9, s = row & 511;
  bool valid = s < lens[b];
  const float* src = attended + (long long)row * D2;
  float acc = 0.0f;
#pragma unroll
  for (int i = 0; i < 16; i++) { int k = lane + i * 32; acc += src[k] * Wp[k]; }
  for (int off = 16; off; off >>= 1) acc += __shfl_xor(acc, off, 32);
  if (lane == 0) out[row] = sigm((valid ? acc : 0.0f) + bp[0]);
}

// ---------------- driver ----------------
extern "C" void kernel_launch(void* const* d_in, const int* in_sizes, int n_in,
                              void* d_out, int out_size, void* d_ws, size_t ws_size,
                              hipStream_t stream) {
  (void)in_sizes; (void)n_in; (void)out_size; (void)ws_size;
  const int*   tok   = (const int*)d_in[0];
  const int*   lens  = (const int*)d_in[1];
  const float* sent  = (const float*)d_in[2];
  const float* emb   = (const float*)d_in[3];
  const float* Wih[4]; const float* Whh[4]; const float* bih[4]; const float* bhh[4];
  for (int i = 0; i < 4; i++) {
    Wih[i] = (const float*)d_in[4 + 4 * i + 0];
    Whh[i] = (const float*)d_in[4 + 4 * i + 1];
    bih[i] = (const float*)d_in[4 + 4 * i + 2];
    bhh[i] = (const float*)d_in[4 + 4 * i + 3];
  }
  const float* Wq = (const float*)d_in[20];
  const float* bq = (const float*)d_in[21];
  const float* Wp = (const float*)d_in[22];
  const float* bp = (const float*)d_in[23];
  float* out = (float*)d_out;

  // workspace layout (bytes), ~262 MB with reuse:
  //  qin overlays x0+x1 (dead after layer-1 LSTM); attended overlays q+scores.
  char* ws = (char*)d_ws;
  unsigned short* x0    = (unsigned short*)(ws + 0);            // 32768*320*2  = 20971520
  unsigned short* x1    = (unsigned short*)(ws + 20971520);     // 32768*512*2  = 33554432
  unsigned short* qin   = (unsigned short*)(ws + 0);            // 32768*576*2  = 37748736 (reuse)
  unsigned short* enc   = (unsigned short*)(ws + 54525952);     // 33554432
  unsigned short* encT  = (unsigned short*)(ws + 88080384);     // 33554432
  unsigned short* qb    = (unsigned short*)(ws + 121634816);    // 33554432
  float*          sc    = (float*)(ws + 155189248);             // 64*512*512*4 = 67108864
  unsigned short* attnb = (unsigned short*)(ws + 222298112);    // 33554432
  float*          attd  = (float*)(ws + 121634816);             // 67108864 (reuse q+scores)
  unsigned short* W2l0  = (unsigned short*)(ws + 255852544);    // 2*1024*576*2 = 2359296
  unsigned short* W2l1  = (unsigned short*)(ws + 258211840);    // 2*1024*768*2 = 3145728
  unsigned short* WqB   = (unsigned short*)(ws + 261357568);    // 512*576*2    = 589824
  float*          bias0 = (float*)(ws + 261947392);             // 2*1024*4
  float*          bias1 = (float*)(ws + 261955584);             // 2*1024*4

  // --- weight prep ---
  {
    dim3 g0((576 + 255) / 256, NG);
    prep_lstm_w_kernel<<<g0, 256, 0, stream>>>(Wih[0], Whh[0], W2l0,            EE, EP0);
    prep_lstm_w_kernel<<<g0, 256, 0, stream>>>(Wih[1], Whh[1], W2l0 + NG * 576, EE, EP0);
    dim3 g1((768 + 255) / 256, NG);
    prep_lstm_w_kernel<<<g1, 256, 0, stream>>>(Wih[2], Whh[2], W2l1,            D2, D2);
    prep_lstm_w_kernel<<<g1, 256, 0, stream>>>(Wih[3], Whh[3], W2l1 + NG * 768, D2, D2);
    prep_bias_kernel<<<4, 256, 0, stream>>>(bih[0], bhh[0], bias0);
    prep_bias_kernel<<<4, 256, 0, stream>>>(bih[1], bhh[1], bias0 + NG);
    prep_bias_kernel<<<4, 256, 0, stream>>>(bih[2], bhh[2], bias1);
    prep_bias_kernel<<<4, 256, 0, stream>>>(bih[3], bhh[3], bias1 + NG);
    f32_to_bf16_kernel<<<(512 * 576 + 255) / 256, 256, 0, stream>>>(Wq, WqB, 512LL * 576);
  }

  // --- embedding gather (f32 -> bf16, pad 300->320) ---
  embed_kernel<<<(int)((long long)BB * SS * EP0 / 256), 256, 0, stream>>>(tok, emb, x0);

  // --- LSTM layers (persistent, one workgroup per direction, dynamic LDS) ---
  {
    size_t sh0 = (size_t)2 * BB * HH * 2 + (size_t)2 * BB * EP0 * 2;  // 147456 B
    size_t sh1 = (size_t)2 * BB * HH * 2 + (size_t)2 * BB * D2  * 2;  // 196608 B
    lstm_kernel<<<2, 1024, sh0, stream>>>(x0, EP0, EP0, W2l0, bias0, lens, x1);
    lstm_kernel<<<2, 1024, sh1, stream>>>(x1, D2,  D2,  W2l1, bias1, lens, enc);
  }

  // --- attention pipeline ---
  transpose_kernel<<<(int)((long long)BB * D2 * SS / 256), 256, 0, stream>>>(enc, encT);
  qin_kernel<<<(int)((long long)BB * SS * QIND / 256), 256, 0, stream>>>(enc, sent, qin);

  // q = tanh(qin @ Wq^T + bq) -> bf16     M=32768 N=512 K=576
  {
    dim3 g(512 / 64, 32768 / 128, 1);
    gemm_nt_kernel<<<g, 256, 0, stream>>>(qin, QIND, 0, WqB, QIND, 0,
                                          nullptr, qb, D2, 0,
                                          BB * SS, D2, QIND, 1, 1.0f, bq);
  }
  // scores[b] = q[b] @ enc[b]^T           M=N=512 K=512, batched over 64
  {
    dim3 g(512 / 64, 512 / 128, BB);
    gemm_nt_kernel<<<g, 256, 0, stream>>>(qb, D2, (long long)SS * D2,
                                          enc, D2, (long long)SS * D2,
                                          sc, nullptr, SS, (long long)SS * SS,
                                          SS, SS, D2, 0, 1.0f, nullptr);
  }
  // softmax (scale + length mask) -> bf16 attn
  softmax_kernel<<<BB * SS / 8, 256, 0, stream>>>(sc, lens, attnb);
  // attended[b] = attn[b] @ enc[b]        M=512 N=512 K=512 via encT
  {
    dim3 g(512 / 64, 512 / 128, BB);
    gemm_nt_kernel<<<g, 256, 0, stream>>>(attnb, SS, (long long)SS * SS,
                                          encT, SS, (long long)D2 * SS,
                                          attd, nullptr, D2, (long long)SS * D2,
                                          SS, D2, SS, 0, 1.0f, nullptr);
  }
  // prediction = sigmoid(mask(attended) . Wp + bp)
  final_kernel<<<BB * SS / 8, 256, 0, stream>>>(attd, Wp, bp, lens, out);
}
